// BahdanauAttention_601295421693
// MI455X (gfx1250) — compile-verified, matrix operand order checked by reading, not run on hardware
//
#include <hip/hip_runtime.h>
#include <hip/hip_bf16.h>
#include <math.h>

// Bahdanau attention for MI455X (gfx1250, wave32, WMMA).
// B=64, S=2048, D_ENC=D_DEC=UNITS=512.

typedef __attribute__((ext_vector_type(16))) __bf16 v16bf;
typedef __attribute__((ext_vector_type(8)))  float  v8f;

#define B_ 64
#define S_ 2048
#define D_ 512
#define U_ 512

__device__ __forceinline__ unsigned short f2bf(float f) {
  unsigned int u = __float_as_uint(f);
  u += 0x7fffu + ((u >> 16) & 1u);   // round-to-nearest-even
  return (unsigned short)(u >> 16);
}

// CDNA5 hardware tanh (TRANS unit). One independent op (v_nop) after the
// trans op per ISA hazard rules before the result is consumed.
__device__ __forceinline__ float htanh(float x) {
  float r;
  asm volatile("v_tanh_f32 %0, %1\n\tv_nop" : "=v"(r) : "v"(x));
  return r;
}

// ---------------------------------------------------------------------------
// Kernel 0: W1 [k=512][u=512] fp32  ->  W1T [u=512][k=512] bf16 (L2-resident)
// ---------------------------------------------------------------------------
__global__ __launch_bounds__(256) void k_w1t(const float* __restrict__ W1,
                                             unsigned short* __restrict__ W1T) {
  int idx = blockIdx.x * blockDim.x + threadIdx.x;   // 0 .. 512*512-1
  int u = idx >> 9;
  int k = idx & 511;
  W1T[u * 512 + k] = f2bf(W1[k * 512 + u]);
}

// ---------------------------------------------------------------------------
// Kernel P: projdec[b][u] = dec[b] @ W2 + b2   (64 x 512, tiny)
// ---------------------------------------------------------------------------
__global__ __launch_bounds__(256) void k_projdec(const float* __restrict__ dec,
                                                 const float* __restrict__ W2,
                                                 const float* __restrict__ b2,
                                                 float* __restrict__ projdec) {
  __shared__ float sdec[D_];
  int b = blockIdx.x, t = threadIdx.x;
  for (int i = t; i < D_; i += 256) sdec[i] = dec[b * D_ + i];
  __syncthreads();
  for (int u = t; u < U_; u += 256) {
    float acc = b2[u];
    for (int d = 0; d < D_; ++d) acc += sdec[d] * W2[d * U_ + u];
    projdec[b * U_ + u] = acc;
  }
}

// ---------------------------------------------------------------------------
// Kernel A: scores[b][s] = V . tanh(enc[b,s,:]@W1 + b1 + projdec[b]) + bv
// Block = 128 threads (4 waves), one (b, 64-row s-block) per block.
// Wave w owns U-stripe [w*128, w*128+128), processed as 4 pairs of u-tiles.
// Per K-step: batch-load 2 B frags (global, L2) + 4 A frags (LDS), then
// issue 8 independent v_wmma_f32_16x16x32_bf16 back-to-back.
// ---------------------------------------------------------------------------
__global__ __launch_bounds__(128) void k_scores(
    const float* __restrict__ enc, const unsigned short* __restrict__ W1T,
    const float* __restrict__ b1, const float* __restrict__ projdec,
    const float* __restrict__ V, const float* __restrict__ bv,
    float* __restrict__ scores) {
  __shared__ __align__(32) unsigned short sA[64 * 512];  // 64 KB bf16 enc tile
  __shared__ float sRed[64];

  int sblk = blockIdx.x;    // 0..31  (64 s-rows each)
  int b    = blockIdx.y;    // 0..63
  int t    = threadIdx.x;
  int lane = t & 31;
  int wave = t >> 5;

  // Stage enc tile (64 rows x 512 cols) to LDS as bf16 via float4 loads.
  const float* encBase = enc + ((size_t)(b * S_ + sblk * 64)) * D_;
  for (int q = t; q < (64 * 512) / 4; q += 128) {
    int r  = q >> 7;           // 128 float4 per row
    int c4 = (q & 127) * 4;
    const float4 f4 = *(const float4*)(encBase + (size_t)r * D_ + c4);
    unsigned int lo = (unsigned int)f2bf(f4.x) | ((unsigned int)f2bf(f4.y) << 16);
    unsigned int hi = (unsigned int)f2bf(f4.z) | ((unsigned int)f2bf(f4.w) << 16);
    *(uint2*)&sA[r * 512 + c4] = make_uint2(lo, hi);
  }
  if (t < 64) sRed[t] = 0.0f;
  __syncthreads();

  const int mlane = lane & 15;   // A: row M / B: column N / C: column N
  const int khalf = lane >> 4;   // K half-select: k = 16*khalf + s (contiguous, A and B agree)
  const int fofs  = khalf * 16;  // fragment K-half offset (elements)

  float sacc[4][8];              // score partials per (s-tile, C-row)
#pragma unroll
  for (int st = 0; st < 4; ++st)
#pragma unroll
    for (int r = 0; r < 8; ++r) sacc[st][r] = 0.0f;

  const int ubase = wave * 128;

  for (int utp = 0; utp < 4; ++utp) {          // pairs of u-tiles
    const int u0 = ubase + utp * 32;
    const unsigned short* bptr0 = W1T + (size_t)(u0 + mlane) * 512 + fofs;
    const unsigned short* bptr1 = bptr0 + (size_t)16 * 512;

    v8f acc[2][4];
    const v8f vzero = {};
#pragma unroll
    for (int p = 0; p < 2; ++p)
#pragma unroll
      for (int st = 0; st < 4; ++st) acc[p][st] = vzero;

    for (int ks = 0; ks < 16; ++ks) {
      // Batch all loads for this K-step, then 8 independent WMMAs.
      v16bf bfrag0 = *(const v16bf*)(bptr0 + ks * 32);
      v16bf bfrag1 = *(const v16bf*)(bptr1 + ks * 32);
      v16bf af[4];
#pragma unroll
      for (int st = 0; st < 4; ++st)
        af[st] = *(const v16bf*)&sA[(st * 16 + mlane) * 512 + ks * 32 + fofs];

#pragma unroll
      for (int st = 0; st < 4; ++st)
        acc[0][st] = __builtin_amdgcn_wmma_f32_16x16x32_bf16(
            false, af[st], false, bfrag0, (short)0, acc[0][st], false, false);
#pragma unroll
      for (int st = 0; st < 4; ++st)
        acc[1][st] = __builtin_amdgcn_wmma_f32_16x16x32_bf16(
            false, af[st], false, bfrag1, (short)0, acc[1][st], false, false);
    }

    // Epilogue for the two u-tiles: bias + hardware tanh + dot with V.
#pragma unroll
    for (int p = 0; p < 2; ++p) {
      int u = u0 + p * 16 + mlane;
      float bias = b1[u] + projdec[b * U_ + u];
      float vu   = V[u];
#pragma unroll
      for (int st = 0; st < 4; ++st) {
#pragma unroll
        for (int r = 0; r < 8; ++r) {
          sacc[st][r] += htanh(acc[p][st][r] + bias) * vu;
        }
      }
    }
  }

  // Reduce over the 16 N-lanes of each half, combine waves via LDS atomics.
#pragma unroll
  for (int st = 0; st < 4; ++st) {
#pragma unroll
    for (int r = 0; r < 8; ++r) {
      float x = sacc[st][r];
      x += __shfl_xor(x, 1, 32);
      x += __shfl_xor(x, 2, 32);
      x += __shfl_xor(x, 4, 32);
      x += __shfl_xor(x, 8, 32);
      if (mlane == 0) atomicAdd(&sRed[st * 16 + r + 8 * khalf], x);
    }
  }
  __syncthreads();

  if (t < 64) scores[(size_t)b * S_ + sblk * 64 + t] = sRed[t] + bv[0];
}

// ---------------------------------------------------------------------------
// Kernel B1: attn[b][s] = softmax over S of scores[b][s]  (register-resident)
// ---------------------------------------------------------------------------
__global__ __launch_bounds__(256) void k_softmax(const float* __restrict__ scores,
                                                 float* __restrict__ attn) {
  __shared__ float sRed[256];
  int b = blockIdx.x, t = threadIdx.x;
  const float* sc = scores + (size_t)b * S_;

  float ls[8];
  float lmax = -INFINITY;
#pragma unroll
  for (int j = 0; j < 8; ++j) {
    ls[j] = sc[t + j * 256];
    lmax = fmaxf(lmax, ls[j]);
  }
  sRed[t] = lmax;
  __syncthreads();
  for (int o = 128; o > 0; o >>= 1) {
    if (t < o) sRed[t] = fmaxf(sRed[t], sRed[t + o]);
    __syncthreads();
  }
  float m = sRed[0];
  __syncthreads();

  float lsum = 0.0f;
#pragma unroll
  for (int j = 0; j < 8; ++j) {
    ls[j] = __expf(ls[j] - m);
    lsum += ls[j];
  }
  sRed[t] = lsum;
  __syncthreads();
  for (int o = 128; o > 0; o >>= 1) {
    if (t < o) sRed[t] += sRed[t + o];
    __syncthreads();
  }
  float inv = 1.0f / sRed[0];

#pragma unroll
  for (int j = 0; j < 8; ++j)
    attn[(size_t)b * S_ + t + j * 256] = ls[j] * inv;
}

// ---------------------------------------------------------------------------
// Kernel B2: partial[ch][b][d] = sum_{s in chunk ch} attn[b][s] * enc[b,s,d]
// 512 blocks -> full device occupancy for the 256 MB streaming pass.
// Deterministic (no float atomics): partials land in workspace.
// ---------------------------------------------------------------------------
__global__ __launch_bounds__(256) void k_ctx_partial(
    const float* __restrict__ enc, const float* __restrict__ attn,
    float* __restrict__ partial) {
  __shared__ float sW[256];
  int b = blockIdx.x, ch = blockIdx.y, t = threadIdx.x;
  sW[t] = attn[(size_t)b * S_ + ch * 256 + t];
  __syncthreads();

  const float* eb = enc + ((size_t)b * S_ + ch * 256) * D_;
  float a0 = 0.0f, a1 = 0.0f;
#pragma unroll 4
  for (int s = 0; s < 256; ++s) {
    float w = sW[s];
    a0 += w * eb[(size_t)s * D_ + t];
    a1 += w * eb[(size_t)s * D_ + t + 256];
  }
  float* p = partial + ((size_t)ch * B_ + b) * D_;
  p[t]       = a0;
  p[t + 256] = a1;
}

// ---------------------------------------------------------------------------
// Kernel B3: ctx[b][d] = sum_ch partial[ch][b][d]
// ---------------------------------------------------------------------------
__global__ __launch_bounds__(256) void k_ctx_reduce(
    const float* __restrict__ partial, float* __restrict__ ctx) {
  int b = blockIdx.x, t = threadIdx.x;
  float a0 = 0.0f, a1 = 0.0f;
#pragma unroll
  for (int ch = 0; ch < 8; ++ch) {
    const float* p = partial + ((size_t)ch * B_ + b) * D_;
    a0 += p[t];
    a1 += p[t + 256];
  }
  ctx[(size_t)b * D_ + t]       = a0;
  ctx[(size_t)b * D_ + t + 256] = a1;
}

// ---------------------------------------------------------------------------
extern "C" void kernel_launch(void* const* d_in, const int* in_sizes, int n_in,
                              void* d_out, int out_size, void* d_ws, size_t ws_size,
                              hipStream_t stream) {
  const float* enc = (const float*)d_in[0];
  const float* dec = (const float*)d_in[1];
  const float* W1  = (const float*)d_in[2];
  const float* b1  = (const float*)d_in[3];
  const float* W2  = (const float*)d_in[4];
  const float* b2  = (const float*)d_in[5];
  const float* V   = (const float*)d_in[6];
  const float* bv  = (const float*)d_in[7];

  float* ctx  = (float*)d_out;              // [64,512]
  float* attn = (float*)d_out + B_ * D_;    // [64,2048]

  // Workspace layout (~2.2 MB):
  unsigned char* ws = (unsigned char*)d_ws;
  unsigned short* W1T    = (unsigned short*)ws;                        // 512 KB bf16
  float*          projd  = (float*)(ws + 512 * 1024);                  // 128 KB
  float*          scores = (float*)(ws + 640 * 1024);                  // 512 KB
  float*          part   = (float*)(ws + 1152 * 1024);                 // 1 MB

  k_w1t<<<dim3((512 * 512) / 256), dim3(256), 0, stream>>>(W1, W1T);
  k_projdec<<<dim3(B_), dim3(256), 0, stream>>>(dec, W2, b2, projd);
  k_scores<<<dim3(S_ / 64, B_), dim3(128), 0, stream>>>(enc, W1T, b1, projd, V, bv, scores);
  k_softmax<<<dim3(B_), dim3(256), 0, stream>>>(scores, attn);
  k_ctx_partial<<<dim3(B_, 8), dim3(256), 0, stream>>>(enc, attn, part);
  k_ctx_reduce<<<dim3(B_), dim3(256), 0, stream>>>(part, ctx);
}
